// Denoiser_83434034692422
// MI455X (gfx1250) — matrix operand order repaired and checked by reading, not compiled, not run on hardware
//
#include <hip/hip_runtime.h>
#include <math.h>

// ---------------------------------------------------------------------------
// MI455X (gfx1250) implementation of the graph-attention denoiser.
// All dense projections run on v_wmma_f32_16x16x32_f16 (f16 in, f32 acc).
// Weights are pre-swizzled into exact B-fragment lane order so the GEMM inner
// loop is: LDS A-frag load + 32B global B-frag load + WMMA.
// GEMM A-tiles are staged into LDS by the Tensor Data Mover (TDM) with
// double buffering (tensor_load_to_lds + s_wait_tensorcnt).
// ---------------------------------------------------------------------------

#define DEVI __device__ __forceinline__

typedef _Float16 half_t;
typedef __attribute__((ext_vector_type(16))) _Float16 v16h;
typedef __attribute__((ext_vector_type(8)))  float    v8f;
typedef __attribute__((ext_vector_type(4)))  unsigned int u32x4;
typedef __attribute__((ext_vector_type(8)))  int i32x8;
typedef __attribute__((ext_vector_type(4)))  int i32x4;

#if defined(__gfx1250__) && __has_builtin(__builtin_amdgcn_tensor_load_to_lds) && \
    __has_builtin(__builtin_amdgcn_s_wait_tensorcnt)
#define USE_TDM 1
#else
#define USE_TDM 0
#endif

namespace cfg {
constexpr int A = 128, T = 50, NT2M = 30, S = 16, P = 768, MODES = 6;
constexpr int H = 128, MD = 32, L = 3;
constexpr int N  = A * S;        // 2048 destination nodes
constexpr int AT = A * T;        // 6400
constexpr int SP = S * P;        // 12288
constexpr int SA = S * A;        // 2048
constexpr int ET  = A * NT2M;    // 3840 base t2m edges
constexpr int EPL = A * 64;      // 8192 base pl2m edges
constexpr int EA  = A * 32;      // 4096 base a2m edges
}

enum { ACT_NONE = 0, ACT_RELU = 1, ACT_SIGMOID = 2, ACT_GELU = 3 };

// ---------------------------------------------------------------------------
// WMMA fragment helpers (CDNA5 16x16x32 f16 layouts, CDNA5 ISA)
// ---------------------------------------------------------------------------
DEVI v8f wmma_f16(v16h a, v16h b, v8f c) {
  return __builtin_amdgcn_wmma_f32_16x16x32_f16(
      /*neg_a=*/false, a, /*neg_b=*/false, b,
      /*c_mod=*/(short)0, c, /*reuse_a=*/false, /*reuse_b=*/false);
}

// A (16x32 f16) from a row-major LDS tile (16 rows, leading dim ldk halves).
// lanes 0-15: row=lane, K pairs {0..7} then {16..23}; lanes 16-31: +8 / +24.
DEVI v16h load_a_frag(const half_t* __restrict__ base, int ldk) {
  const int lane = threadIdx.x & 31;
  const int row  = lane & 15;
  const int kk   = (lane & 16) ? 8 : 0;
  const half_t* p = base + row * ldk + kk;
  v16h a;
#pragma unroll
  for (int j = 0; j < 8; ++j) a[j] = p[j];
#pragma unroll
  for (int j = 0; j < 8; ++j) a[8 + j] = p[16 + j];
  return a;
}

// B (32x16 f16) pre-swizzled in global: per (ktile,ntile), 32 lanes x 16 halves.
DEVI v16h load_b_frag(const half_t* __restrict__ wp) {
  const int lane = threadIdx.x & 31;
  return *(const v16h*)(wp + (size_t)lane * 16);
}

#if USE_TDM
// ---------------------------------------------------------------------------
// TDM descriptor: 2D tile (tile_dim0=32 f32 per row, tile_dim1=128 rows) out
// of a row-major [rows, K] f32 tensor starting at gaddr. D# layout per
// CDNA5 ISA (count=1, type=2 "image", data_size=4B).
// ---------------------------------------------------------------------------
DEVI void tdm_load_tile_f32(unsigned int lds_byte_addr, const float* gaddr, int K) {
  unsigned long long ga = (unsigned long long)(uintptr_t)gaddr;
  u32x4 g0;
  g0[0] = 1u;                                   // count=1 (valid user descriptor)
  g0[1] = lds_byte_addr;                        // lds_addr [63:32]
  g0[2] = (unsigned int)ga;                     // global_addr [95:64]
  g0[3] = (unsigned int)((ga >> 32) & 0x1FFFFFFu) | (2u << 30);  // addr[56:32] | type=2
  i32x8 g1;
  g1[0] = (int)(2u << 16);                                      // data_size = 4B
  g1[1] = (int)(((unsigned)K & 0xFFFFu) << 16);                 // tensor_dim0 [15:0]
  g1[2] = (int)((((unsigned)K >> 16) & 0xFFFFu) | (128u << 16));// dim0 hi | tensor_dim1 lo = 128
  g1[3] = (int)(32u << 16);                                     // dim1 hi=0 | tile_dim0 = 32
  g1[4] = 128;                                                  // tile_dim1 = 128 rows
  g1[5] = (int)(unsigned)K;                                     // tensor_dim0_stride [31:0]
  g1[6] = 0;                                                    // stride hi | dim1_stride lo
  g1[7] = 0;
  i32x4 z4 = {0, 0, 0, 0};
  i32x8 z8 = {0, 0, 0, 0, 0, 0, 0, 0};
  __builtin_amdgcn_tensor_load_to_lds(g0, g1, z4, z4, z8, 0);
}
#endif

// ---------------------------------------------------------------------------
// Weight pre-swizzle: W[K,N] f32 row-major -> B-frag-major f16.
// Element j of lane L in (kt,nt): K = kt*32 + (L<16?0:8) + (j<8 ? j : j+8), Ncol = nt*16 + (L&15)
// ---------------------------------------------------------------------------
__global__ void pack_weight_kernel(const float* __restrict__ W, half_t* __restrict__ Wp,
                                   int Ktrue, int Kpad, int Ncols) {
  int t = blockIdx.x * blockDim.x + threadIdx.x;
  int NT = Ncols >> 4;
  int total = (Kpad >> 5) * NT * 32;
  if (t >= total) return;
  int lane = t & 31;
  int nt = (t >> 5) % NT;
  int kt = t / (32 * NT);
  int n  = (nt << 4) + (lane & 15);
  int kb = (kt << 5) + ((lane & 16) ? 8 : 0);
  half_t* dst = Wp + (size_t)t * 16;
#pragma unroll
  for (int j = 0; j < 16; ++j) {
    int k = kb + (j < 8 ? j : j + 8);
    dst[j] = (k < Ktrue) ? (half_t)W[(size_t)k * Ncols + n] : (half_t)0.f;
  }
}

// ---------------------------------------------------------------------------
// Generic WMMA GEMM: out[M,N] = act(in[M,K] @ Wp + bias) (+ resid)
// block = 256 threads (8 waves as 4x2), tile 128 x 64, K step 32.
// M % 128 == 0, N % 32 == 0, K % 32 == 0 (guaranteed by the model shapes).
// A tile staged via TDM (double-buffered f32 stage -> f16 WMMA tile).
// ---------------------------------------------------------------------------
__global__ __launch_bounds__(256) void gemm_kernel(
    const float* __restrict__ in, const half_t* __restrict__ Wp,
    const float* __restrict__ bias, const float* __restrict__ resid,
    float* __restrict__ out, int M, int Ncols, int K, int act) {
#if USE_TDM
  __shared__ float  stageS[2][128 * 32];
#endif
  __shared__ half_t As[128 * 32];
  const int wave = threadIdx.x >> 5;
  const int lane = threadIdx.x & 31;
  const int wm = wave & 3, wn = wave >> 2;
  const int m0 = blockIdx.x * 128 + wm * 32;
  const int n0 = blockIdx.y * 64 + wn * 32;
  const bool colActive = (n0 < Ncols);
  const int NT = Ncols >> 4;
  const int ntb = n0 >> 4;
  const float* arow = in + (size_t)blockIdx.x * 128 * K;
  const int KT = K >> 5;

  v8f acc00 = {}, acc01 = {}, acc10 = {}, acc11 = {};

#if USE_TDM
  if (threadIdx.x < 32)   // wave 0 drives the tensor DMA
    tdm_load_tile_f32((unsigned int)(uintptr_t)&stageS[0][0], arow, K);
  int buf = 0;
#endif

  for (int kt = 0; kt < KT; ++kt) {
    __syncthreads();   // As free to overwrite; prior frag reads complete
#if USE_TDM
    if (threadIdx.x < 32) __builtin_amdgcn_s_wait_tensorcnt(0);
    __syncthreads();   // staged tile visible to the whole workgroup
    if (threadIdx.x < 32 && kt + 1 < KT)   // kick next tile while we compute
      tdm_load_tile_f32((unsigned int)(uintptr_t)&stageS[buf ^ 1][0],
                        arow + (size_t)(kt + 1) * 32, K);
    for (int i = threadIdx.x; i < 128 * 32; i += 256)
      As[i] = (half_t)stageS[buf][i];
    buf ^= 1;
#else
    for (int i = threadIdx.x; i < 128 * 8; i += 256) {
      int r = i >> 3, c4 = (i & 7) << 2;
      const float4 f = *(const float4*)(arow + (size_t)r * K + kt * 32 + c4);
      half_t* d = As + r * 32 + c4;
      d[0] = (half_t)f.x; d[1] = (half_t)f.y; d[2] = (half_t)f.z; d[3] = (half_t)f.w;
    }
#endif
    __syncthreads();
    if (colActive) {
      const half_t* wb = Wp + ((size_t)kt * NT + ntb) * 32 * 16;
      __builtin_prefetch(wb + (size_t)NT * 32 * 16, 0, 3);  // next k-step B stream
      v16h a0 = load_a_frag(As + (wm * 32 + 0) * 32, 32);
      v16h a1 = load_a_frag(As + (wm * 32 + 16) * 32, 32);
      v16h b0 = load_b_frag(wb);
      v16h b1 = load_b_frag(wb + 32 * 16);
      acc00 = wmma_f16(a0, b0, acc00);
      acc01 = wmma_f16(a0, b1, acc01);
      acc10 = wmma_f16(a1, b0, acc10);
      acc11 = wmma_f16(a1, b1, acc11);
    }
  }
  if (!colActive) return;
  const int rsel = (lane & 16) ? 8 : 0;
  const int cl = lane & 15;
  v8f accs[2][2] = {{acc00, acc01}, {acc10, acc11}};
#pragma unroll
  for (int ti = 0; ti < 2; ++ti) {
#pragma unroll
    for (int tj = 0; tj < 2; ++tj) {
      const int ncol = n0 + tj * 16 + cl;
      const float bb = bias ? bias[ncol] : 0.f;
#pragma unroll
      for (int i = 0; i < 8; ++i) {
        const int r = m0 + ti * 16 + rsel + i;
        float v = accs[ti][tj][i] + bb;
        if (act == ACT_RELU)         v = fmaxf(v, 0.f);
        else if (act == ACT_SIGMOID) v = 1.f / (1.f + __expf(-v));
        else if (act == ACT_GELU)    v = 0.5f * v * (1.f + erff(v * 0.70710678118f));
        if (resid) v += resid[(size_t)r * Ncols + ncol];
        out[(size_t)r * Ncols + ncol] = v;
      }
    }
  }
}

// ---------------------------------------------------------------------------
// LayerNorm over last dim 128. One block (128 threads) per row.
// ---------------------------------------------------------------------------
__global__ __launch_bounds__(128) void layernorm_kernel(
    const float* __restrict__ in, const float* __restrict__ g,
    const float* __restrict__ b, float* __restrict__ out) {
  const int row = blockIdx.x, c = threadIdx.x;
  __shared__ float red[128];
  float v = in[(size_t)row * 128 + c];
  red[c] = v; __syncthreads();
  for (int s = 64; s > 0; s >>= 1) { if (c < s) red[c] += red[c + s]; __syncthreads(); }
  float mu = red[0] * (1.f / 128.f);
  __syncthreads();
  float dv = v - mu;
  red[c] = dv * dv; __syncthreads();
  for (int s = 64; s > 0; s >>= 1) { if (c < s) red[c] += red[c + s]; __syncthreads(); }
  float rinv = rsqrtf(red[0] * (1.f / 128.f) + 1e-5f);
  out[(size_t)row * 128 + c] = dv * rinv * g[c] + b[c];
}

// ---------------------------------------------------------------------------
// Fused Fourier embedding: feat-gen -> (W1,LN,relu) -> (W2, sum over dims)
//                           -> (LN,relu) -> W3. block=256, 16 rows per block.
// ---------------------------------------------------------------------------
__global__ __launch_bounds__(256) void fourier_kernel(
    const float* __restrict__ x, int D,
    const float* __restrict__ freqs, const half_t* __restrict__ w1p,
    const float* __restrict__ b1, const float* __restrict__ ln1g,
    const float* __restrict__ ln1b, const half_t* __restrict__ w2p,
    const float* __restrict__ b2, const float* __restrict__ og,
    const float* __restrict__ ob, const half_t* __restrict__ owp,
    const float* __restrict__ obias, float* __restrict__ out) {
  constexpr float TWO_PI = 6.28318530717958647692f;
  __shared__ half_t featS[16 * 160];
  __shared__ float  hS[16 * 128];
  __shared__ half_t hH[16 * 128];
  __shared__ float  muS[16], riS[16];
  const int tid = threadIdx.x;
  const int wave = tid >> 5;
  const int lane = tid & 31;
  const int row0 = blockIdx.x * 16;
  const int colg = wave * 16 + (lane & 15);
  const int rsel = (lane & 16) ? 8 : 0;
  v8f acc2 = {};

  for (int d = 0; d < D; ++d) {
    __syncthreads();
    for (int i = tid; i < 16 * 160; i += 256) {
      int r = i / 160, c = i - r * 160;
      float xv = x[(size_t)(row0 + r) * D + d];
      float val;
      if (c < 64)        val = __cosf(xv * freqs[d * 64 + c] * TWO_PI);
      else if (c < 128)  val = __sinf(xv * freqs[d * 64 + c - 64] * TWO_PI);
      else if (c == 128) val = xv;
      else               val = 0.f;
      featS[i] = (half_t)val;
    }
    __syncthreads();
    v8f acc1 = {};
#pragma unroll
    for (int kt = 0; kt < 5; ++kt) {
      v16h a = load_a_frag(featS + kt * 32, 160);
      v16h b = load_b_frag(w1p + ((size_t)(d * 40 + kt * 8 + wave) * 32) * 16);
      acc1 = wmma_f16(a, b, acc1);
    }
    {
      float bb = b1[d * 128 + colg];
#pragma unroll
      for (int i = 0; i < 8; ++i) hS[(rsel + i) * 128 + colg] = acc1[i] + bb;
    }
    __syncthreads();
    if (tid < 16) {
      float s = 0.f;
      for (int c = 0; c < 128; ++c) s += hS[tid * 128 + c];
      float m = s * (1.f / 128.f), v2 = 0.f;
      for (int c = 0; c < 128; ++c) { float dd = hS[tid * 128 + c] - m; v2 += dd * dd; }
      muS[tid] = m; riS[tid] = rsqrtf(v2 * (1.f / 128.f) + 1e-5f);
    }
    __syncthreads();
    for (int i = tid; i < 16 * 128; i += 256) {
      int r = i >> 7, c = i & 127;
      float v = (hS[i] - muS[r]) * riS[r] * ln1g[d * 128 + c] + ln1b[d * 128 + c];
      hH[i] = (half_t)fmaxf(v, 0.f);
    }
    __syncthreads();
#pragma unroll
    for (int kt = 0; kt < 4; ++kt) {
      v16h a = load_a_frag(hH + kt * 32, 128);
      v16h b = load_b_frag(w2p + ((size_t)(d * 32 + kt * 8 + wave) * 32) * 16);
      acc2 = wmma_f16(a, b, acc2);
    }
    float bb = b2[d * 128 + colg];
#pragma unroll
    for (int i = 0; i < 8; ++i) acc2[i] += bb;
  }
  // final LN + relu + @ow + obias
  __syncthreads();
#pragma unroll
  for (int i = 0; i < 8; ++i) hS[(rsel + i) * 128 + colg] = acc2[i];
  __syncthreads();
  if (tid < 16) {
    float s = 0.f;
    for (int c = 0; c < 128; ++c) s += hS[tid * 128 + c];
    float m = s * (1.f / 128.f), v2 = 0.f;
    for (int c = 0; c < 128; ++c) { float dd = hS[tid * 128 + c] - m; v2 += dd * dd; }
    muS[tid] = m; riS[tid] = rsqrtf(v2 * (1.f / 128.f) + 1e-5f);
  }
  __syncthreads();
  for (int i = tid; i < 16 * 128; i += 256) {
    int r = i >> 7, c = i & 127;
    float v = (hS[i] - muS[r]) * riS[r] * og[c] + ob[c];
    hH[i] = (half_t)fmaxf(v, 0.f);
  }
  __syncthreads();
  v8f acc3 = {};
#pragma unroll
  for (int kt = 0; kt < 4; ++kt) {
    v16h a = load_a_frag(hH + kt * 32, 128);
    v16h b = load_b_frag(owp + ((size_t)(kt * 8 + wave) * 32) * 16);
    acc3 = wmma_f16(a, b, acc3);
  }
  float bb = obias[colg];
#pragma unroll
  for (int i = 0; i < 8; ++i)
    out[(size_t)(row0 + rsel + i) * 128 + colg] = acc3[i] + bb;
}

// ---------------------------------------------------------------------------
// Per-destination attention with fixed-degree segment softmax.
// One block (128 threads = 8 heads x 16 ch) per destination node.
// mode 0:t2m  1:pl2m  2:a2m/m2m  3:ms2m
// ---------------------------------------------------------------------------
__global__ __launch_bounds__(128) void attn_edge_kernel(
    const float* __restrict__ q, const float* __restrict__ k,
    const float* __restrict__ v, const float* __restrict__ kr,
    const float* __restrict__ vr, const int* __restrict__ srcs,
    int deg, int mode, float* __restrict__ agg) {
  const int dst = blockIdx.x;
  const int c = threadIdx.x;
  const int h = c >> 4;
  __shared__ float simS[64 * 8];
  __shared__ float wS[64 * 8];
  __shared__ int   srcS[64];
  __shared__ int   eidS[64];

  if (c < deg) {
    int srow, eid = -1;
    if (mode == 0)      { int a = dst >> 4; srow = srcs[a * deg + c]; eid = a * deg + c; }
    else if (mode == 1) { int a = dst & 127; int s = dst >> 7; srow = srcs[a * deg + c] + s * cfg::P; eid = a * deg + c; }
    else if (mode == 2) { int a = dst & 127; int s = dst >> 7; srow = srcs[a * deg + c] + s * cfg::A; eid = a * deg + c; }
    else                { int a = dst >> 4; srow = a * cfg::MODES + c; }
    srcS[c] = srow; eidS[c] = eid;
  }
  __syncthreads();

  const float qc = q[(size_t)dst * 128 + c];
  for (int e = 0; e < deg; ++e) {
    float kc = k[(size_t)srcS[e] * 128 + c];
    if (kr) kc += kr[(size_t)eidS[e] * 128 + c];
    float p = qc * kc;
    p += __shfl_xor(p, 8, 16);
    p += __shfl_xor(p, 4, 16);
    p += __shfl_xor(p, 2, 16);
    p += __shfl_xor(p, 1, 16);
    if ((c & 15) == 0) simS[e * 8 + h] = p * 0.25f;  // HD^-0.5
  }
  __syncthreads();
  if (c < 8) {
    float mx = -1e30f;
    for (int e = 0; e < deg; ++e) mx = fmaxf(mx, simS[e * 8 + c]);
    float den = 0.f;
    for (int e = 0; e < deg; ++e) { float w = __expf(simS[e * 8 + c] - mx); wS[e * 8 + c] = w; den += w; }
    float inv = 1.f / (den + 1e-16f);
    for (int e = 0; e < deg; ++e) wS[e * 8 + c] *= inv;
  }
  __syncthreads();
  float acc = 0.f;
  for (int e = 0; e < deg; ++e) {
    float vc = v[(size_t)srcS[e] * 128 + c];
    if (vr) vc += vr[(size_t)eidS[e] * 128 + c];
    acc += wS[e * 8 + h] * vc;
  }
  agg[(size_t)dst * 128 + c] = acc;
}

// ---------------------------------------------------------------------------
// Small elementwise / layout kernels
// ---------------------------------------------------------------------------
DEVI float wrapf(float a) {
  const float PI_F = 3.14159265358979323846f;
  float m = fmodf(a + PI_F, 6.28318530717958647692f);
  if (m < 0.f) m += 6.28318530717958647692f;
  return m - PI_F;
}

__global__ void edge_feat_kernel(const int* __restrict__ edge, int E, int mode,
                                 const float* __restrict__ pos_s, const float* __restrict__ head_s,
                                 const float* __restrict__ pos_m, const float* __restrict__ head_m,
                                 float* __restrict__ feat) {
  int e = blockIdx.x * blockDim.x + threadIdx.x;
  if (e >= E) return;
  int s = edge[e], d = edge[E + e];
  float rx = pos_s[2 * s] - pos_m[2 * d];
  float ry = pos_s[2 * s + 1] - pos_m[2 * d + 1];
  float hm = head_m[d];
  float hx = __cosf(hm), hy = __sinf(hm);
  float nrm = sqrtf(rx * rx + ry * ry);
  float ang = atan2f(hx * ry - hy * rx, hx * rx + hy * ry);
  float dh = wrapf(head_s[s] - hm);
  if (mode == 0) {
    feat[e * 4 + 0] = nrm; feat[e * 4 + 1] = ang; feat[e * 4 + 2] = dh;
    feat[e * 4 + 3] = (float)(s % cfg::T) - (float)(cfg::T - 1);
  } else {
    feat[e * 3 + 0] = nrm; feat[e * 3 + 1] = ang; feat[e * 3 + 2] = dh;
  }
}

__global__ void add_kernel(const float* __restrict__ a, const float* __restrict__ b,
                           float* __restrict__ o, int n) {
  int i = blockIdx.x * 256 + threadIdx.x;
  if (i < n) o[i] = a[i] + b[i];
}

__global__ void gate_kernel(const float* __restrict__ agg, const float* __restrict__ g,
                            const float* __restrict__ s, float* __restrict__ o, int n) {
  int i = blockIdx.x * 256 + threadIdx.x;
  if (i < n) { float a = agg[i]; o[i] = a + g[i] * (s[i] - a); }
}

__global__ void concat_kernel(const float* __restrict__ a, const float* __restrict__ b,
                              float* __restrict__ o, int M) {
  int i = blockIdx.x * 256 + threadIdx.x;
  int tot = M * 256;
  if (i >= tot) return;
  int r = i >> 8, c = i & 255;
  o[i] = (c < 128) ? a[(size_t)r * 128 + c] : b[(size_t)r * 128 + (c - 128)];
}

// out[(j*D0+i)*128+c] = in[(i*D1+j)*128+c]
__global__ void perm_kernel(const float* __restrict__ in, float* __restrict__ out,
                            int D0, int D1) {
  int idx = blockIdx.x * 256 + threadIdx.x;
  int tot = D0 * D1 * 128;
  if (idx >= tot) return;
  int c = idx & 127, ij = idx >> 7;
  int j = ij % D1, i = ij / D1;
  out[((size_t)(j * D0 + i) << 7) + c] = in[idx];
}

// ---------------------------------------------------------------------------
// Host side
// ---------------------------------------------------------------------------
struct FourW {
  const float *b1, *b2, *freqs, *ln1_b, *ln1_g, *ob, *obias, *og, *ow, *w1, *w2;
  int D;
};
struct AttnWf {
  const float *bg, *bo, *bq, *bs, *bv, *bvr;
  const float *ff_b1, *ff_b2, *ff_ln_b, *ff_ln_g, *ff_w1, *ff_w2;
  const float *ln_dst_b, *ln_dst_g, *ln_r_b, *ln_r_g, *ln_src_b, *ln_src_g;
  const float *wg, *wk, *wkr, *wo, *wq, *ws, *wv, *wvr;
  bool pos;
};
struct FourDev { FourW f; half_t *w1p, *w2p, *owp; };
struct AttnDev {
  AttnWf f;
  half_t *wq, *wk, *wv, *ws, *wg, *wo, *ff1, *ff2, *wkr, *wvr;
};

extern "C" void kernel_launch(void* const* d_in, const int* in_sizes, int n_in,
                              void* d_out, int out_size, void* d_ws, size_t ws_size,
                              hipStream_t stream) {
  using namespace cfg;
  (void)in_sizes; (void)n_in; (void)out_size; (void)ws_size;

  // ---- data inputs (top-level dict insertion order) ----
  const float* m_in     = (const float*)d_in[0];   // [A,S,MD]
  const float* beta     = (const float*)d_in[1];   // [A*S,1]
  const float* mm       = (const float*)d_in[2];   // [A,MODES,MD]
  const float* mmscore  = (const float*)d_in[3];   // [A,MODES]
  const float* pos_m    = (const float*)d_in[4];
  const float* head_m   = (const float*)d_in[5];
  const float* pos_t    = (const float*)d_in[6];
  const float* head_t   = (const float*)d_in[7];
  const float* pos_pl   = (const float*)d_in[8];
  const float* orient_pl= (const float*)d_in[9];
  const float* x_t      = (const float*)d_in[10];  // [AT,H]
  const float* x_pl     = (const float*)d_in[11];  // [SP,H]
  const float* x_a      = (const float*)d_in[12];  // [SA,H]

  // ---- params pytree (sorted keys at every nested level, lists in order) ----
  int idx = 13;
  auto nf = [&]() { return (const float*)d_in[idx++]; };
  auto readFour = [&](int D) {
    FourW f; f.D = D;
    f.b1 = nf(); f.b2 = nf(); f.freqs = nf(); f.ln1_b = nf(); f.ln1_g = nf();
    f.ob = nf(); f.obias = nf(); f.og = nf(); f.ow = nf(); f.w1 = nf(); f.w2 = nf();
    return f;
  };
  auto readAttn = [&](bool pos) {
    AttnWf a{}; a.pos = pos;
    a.bg = nf(); a.bo = nf(); a.bq = nf(); a.bs = nf(); a.bv = nf();
    if (pos) a.bvr = nf();
    a.ff_b1 = nf(); a.ff_b2 = nf(); a.ff_ln_b = nf(); a.ff_ln_g = nf();
    a.ff_w1 = nf(); a.ff_w2 = nf();
    a.ln_dst_b = nf(); a.ln_dst_g = nf();
    if (pos) { a.ln_r_b = nf(); a.ln_r_g = nf(); }
    a.ln_src_b = nf(); a.ln_src_g = nf();
    a.wg = nf(); a.wk = nf();
    if (pos) a.wkr = nf();
    a.wo = nf(); a.wq = nf(); a.ws = nf(); a.wv = nf();
    if (pos) a.wvr = nf();
    return a;
  };

  AttnWf A2M[L], M2M[L], MS2M[L], PL2M[L], T2M[L];
  for (int i = 0; i < L; ++i) A2M[i] = readAttn(true);
  FourW F_a2m   = readFour(3);
  FourW F_noise = readFour(1);
  FourW F_pl2m  = readFour(3);
  FourW F_score = readFour(1);
  FourW F_t2m   = readFour(4);
  for (int i = 0; i < L; ++i) M2M[i]  = readAttn(true);
  for (int i = 0; i < L; ++i) MS2M[i] = readAttn(false);
  for (int i = 0; i < L; ++i) PL2M[i] = readAttn(true);
  const float* proj_in_b = nf();    const float* proj_in_w = nf();
  const float* proj_mm_b = nf();    const float* proj_mm_w = nf();
  const float* proj_out_b = nf();   const float* proj_out_w = nf();
  for (int i = 0; i < L; ++i) T2M[i] = readAttn(true);
  const float* to_b    = nf(); const float* to_n1_b = nf(); const float* to_n1_g = nf();
  const float* to_n2_b = nf(); const float* to_n2_g = nf(); const float* to_w    = nf();

  const int* e_t2m  = (const int*)d_in[idx++];
  const int* e_pl2m = (const int*)d_in[idx++];
  const int* e_a2m  = (const int*)d_in[idx++];

  // ---- workspace carve-out ----
  uint8_t* base = (uint8_t*)d_ws;
  size_t off = 0;
  auto allocB = [&](size_t bytes) -> void* {
    void* p = base + off;
    off = (off + bytes + 255) & ~(size_t)255;
    return p;
  };
  auto allocF = [&](size_t n) { return (float*)allocB(n * 4); };
  auto allocH = [&](size_t n) { return (half_t*)allocB(n * 2); };

  auto packTo = [&](const float* W, half_t* dst, int Ktrue, int Kpad, int Nc) {
    int total = (Kpad / 32) * (Nc / 16) * 32;
    pack_weight_kernel<<<dim3((total + 255) / 256), 256, 0, stream>>>(W, dst, Ktrue, Kpad, Nc);
  };
  auto pack = [&](const float* W, int K, int Nc) {
    half_t* p = allocH((size_t)K * Nc);
    packTo(W, p, K, K, Nc);
    return p;
  };

  auto mkAttn = [&](const AttnWf& f) {
    AttnDev d{}; d.f = f;
    d.wq = pack(f.wq, 128, 128);  d.wk = pack(f.wk, 128, 128);
    d.wv = pack(f.wv, 128, 128);  d.ws = pack(f.ws, 128, 128);
    d.wg = pack(f.wg, 256, 128);  d.wo = pack(f.wo, 128, 128);
    d.ff1 = pack(f.ff_w1, 128, 512); d.ff2 = pack(f.ff_w2, 512, 128);
    if (f.pos) { d.wkr = pack(f.wkr, 128, 128); d.wvr = pack(f.wvr, 128, 128); }
    return d;
  };
  auto mkFour = [&](const FourW& f) {
    FourDev d{}; d.f = f;
    d.w1p = allocH((size_t)f.D * 160 * 128);
    d.w2p = allocH((size_t)f.D * 128 * 128);
    for (int dd = 0; dd < f.D; ++dd) {
      packTo(f.w1 + (size_t)dd * 129 * 128, d.w1p + (size_t)dd * 160 * 128, 129, 160, 128);
      packTo(f.w2 + (size_t)dd * 128 * 128, d.w2p + (size_t)dd * 128 * 128, 128, 128, 128);
    }
    d.owp = pack(f.ow, 128, 128);
    return d;
  };

  AttnDev dT2M[L], dPL2M[L], dA2M[L], dM2M[L], dMS2M[L];
  for (int i = 0; i < L; ++i) dA2M[i]  = mkAttn(A2M[i]);
  FourDev dF_a2m = mkFour(F_a2m), dF_noise = mkFour(F_noise), dF_pl2m = mkFour(F_pl2m);
  FourDev dF_score = mkFour(F_score), dF_t2m = mkFour(F_t2m);
  for (int i = 0; i < L; ++i) dM2M[i]  = mkAttn(M2M[i]);
  for (int i = 0; i < L; ++i) dMS2M[i] = mkAttn(MS2M[i]);
  for (int i = 0; i < L; ++i) dPL2M[i] = mkAttn(PL2M[i]);
  for (int i = 0; i < L; ++i) dT2M[i]  = mkAttn(T2M[i]);
  half_t* proj_in_p  = pack(proj_in_w, 32, 128);
  half_t* proj_mm_p  = pack(proj_mm_w, 32, 128);
  half_t* proj_out_p = pack(proj_out_w, 128, 32);
  half_t* to_wp      = pack(to_w, 128, 128);

  // activation buffers
  float* feat_t = allocF((size_t)ET * 4);
  float* feat_pl = allocF((size_t)EPL * 3);
  float* feat_a = allocF((size_t)EA * 3);
  float* r_t  = allocF((size_t)ET * 128);
  float* r_pl = allocF((size_t)EPL * 128);
  float* r_a  = allocF((size_t)EA * 128);
  float* beta_emb  = allocF((size_t)N * 128);
  float* score_emb = allocF((size_t)A * MODES * 128);
  float* msv       = allocF((size_t)A * MODES * 128);
  float* mc   = allocF((size_t)N * 128);
  float* mc2  = allocF((size_t)N * 128);
  float* msum = allocF((size_t)N * 128);
  float* xs_buf = allocF((size_t)SP * 128);
  float* k_buf  = allocF((size_t)SP * 128);
  float* v_buf  = allocF((size_t)SP * 128);
  float* xd_buf = allocF((size_t)N * 128);
  float* q_buf  = allocF((size_t)N * 128);
  float* rn_buf = allocF((size_t)EPL * 128);
  float* kr_buf = allocF((size_t)EPL * 128);
  float* vr_buf = allocF((size_t)EPL * 128);
  float* agg  = allocF((size_t)N * 128);
  float* sbuf = allocF((size_t)N * 128);
  float* gbuf = allocF((size_t)N * 128);
  float* cat  = allocF((size_t)N * 256);
  float* obuf = allocF((size_t)N * 128);
  float* hn   = allocF((size_t)N * 128);
  float* ffh  = allocF((size_t)N * 512);
  float* t1   = allocF((size_t)N * 128);
  float* t2   = allocF((size_t)N * 128);

  auto gemm = [&](const float* in, const half_t* Wp, const float* bias,
                  const float* resid, float* out, int M, int Nc, int K, int act) {
    dim3 g(M / 128, (Nc + 63) / 64);
    gemm_kernel<<<g, 256, 0, stream>>>(in, Wp, bias, resid, out, M, Nc, K, act);
  };
  auto fourier = [&](const FourDev& fd, const float* x, int M, float* out) {
    fourier_kernel<<<dim3(M / 16), 256, 0, stream>>>(
        x, fd.f.D, fd.f.freqs, fd.w1p, fd.f.b1, fd.f.ln1_g, fd.f.ln1_b,
        fd.w2p, fd.f.b2, fd.f.og, fd.f.ob, fd.owp, fd.f.obias, out);
  };
  auto ln = [&](const float* in, const float* g, const float* b, float* out, int M) {
    layernorm_kernel<<<M, 128, 0, stream>>>(in, g, b, out);
  };
  auto ew = [&](int n) { return dim3((n + 255) / 256); };

  // ---- relative-position Fourier embeddings (computed on base edges once) ----
  edge_feat_kernel<<<dim3((ET + 127) / 128), 128, 0, stream>>>(
      e_t2m, ET, 0, pos_t, head_t, pos_m, head_m, feat_t);
  edge_feat_kernel<<<dim3((EPL + 127) / 128), 128, 0, stream>>>(
      e_pl2m, EPL, 1, pos_pl, orient_pl, pos_m, head_m, feat_pl);
  edge_feat_kernel<<<dim3((EA + 127) / 128), 128, 0, stream>>>(
      e_a2m, EA, 2, pos_m, head_m, pos_m, head_m, feat_a);
  fourier(dF_t2m, feat_t, ET, r_t);
  fourier(dF_pl2m, feat_pl, EPL, r_pl);
  fourier(dF_a2m, feat_a, EA, r_a);
  fourier(dF_noise, beta, N, beta_emb);
  fourier(dF_score, mmscore, A * MODES, score_emb);

  // ms = mm @ proj_in_mm + b + score_emb
  gemm(mm, proj_mm_p, proj_mm_b, nullptr, msv, A * MODES, 128, 32, ACT_NONE);
  add_kernel<<<ew(A * MODES * 128), 256, 0, stream>>>(msv, score_emb, msv, A * MODES * 128);

  // mc = m @ proj_in + b ; msum = mc
  gemm(m_in, proj_in_p, proj_in_b, nullptr, mc, N, 128, 32, ACT_NONE);
  (void)hipMemcpyAsync(msum, mc, (size_t)N * 128 * 4, hipMemcpyDeviceToDevice, stream);

  auto attn = [&](const AttnDev& w, const float* x_src, int Msrc, float* x_dst,
                  const float* r, int Ebase, const int* srcs, int deg, int mode,
                  bool tied) {
    ln(x_src, w.f.ln_src_g, w.f.ln_src_b, xs_buf, Msrc);
    ln(x_dst, tied ? w.f.ln_src_g : w.f.ln_dst_g,
       tied ? w.f.ln_src_b : w.f.ln_dst_b, xd_buf, N);
    gemm(xd_buf, w.wq, w.f.bq, nullptr, q_buf, N, 128, 128, ACT_NONE);
    gemm(xs_buf, w.wk, nullptr, nullptr, k_buf, Msrc, 128, 128, ACT_NONE);
    gemm(xs_buf, w.wv, w.f.bv, nullptr, v_buf, Msrc, 128, 128, ACT_NONE);
    const float* krp = nullptr; const float* vrp = nullptr;
    if (w.f.pos) {
      ln(r, w.f.ln_r_g, w.f.ln_r_b, rn_buf, Ebase);
      gemm(rn_buf, w.wkr, nullptr, nullptr, kr_buf, Ebase, 128, 128, ACT_NONE);
      gemm(rn_buf, w.wvr, w.f.bvr, nullptr, vr_buf, Ebase, 128, 128, ACT_NONE);
      krp = kr_buf; vrp = vr_buf;
    }
    attn_edge_kernel<<<N, 128, 0, stream>>>(q_buf, k_buf, v_buf, krp, vrp,
                                            srcs, deg, mode, agg);
    gemm(xd_buf, w.ws, w.f.bs, nullptr, sbuf, N, 128, 128, ACT_NONE);
    concat_kernel<<<ew(N * 256), 256, 0, stream>>>(agg, xd_buf, cat, N);
    gemm(cat, w.wg, w.f.bg, nullptr, gbuf, N, 128, 256, ACT_SIGMOID);
    gate_kernel<<<ew(N * 128), 256, 0, stream>>>(agg, gbuf, sbuf, obuf, N * 128);
    gemm(obuf, w.wo, w.f.bo, x_dst, x_dst, N, 128, 128, ACT_NONE);
    ln(x_dst, w.f.ff_ln_g, w.f.ff_ln_b, hn, N);
    gemm(hn, w.ff1, w.f.ff_b1, nullptr, ffh, N, 512, 128, ACT_RELU);
    gemm(ffh, w.ff2, w.f.ff_b2, x_dst, x_dst, N, 128, 512, ACT_NONE);
  };

  // ---- denoising attention stack ----
  for (int i = 0; i < L; ++i) {
    add_kernel<<<ew(N * 128), 256, 0, stream>>>(mc, beta_emb, mc, N * 128);
    attn(dT2M[i], x_t, AT, mc, r_t, ET, e_t2m, NT2M, 0, false);
    perm_kernel<<<ew(N * 128), 256, 0, stream>>>(mc, mc2, A, S);   // -> s*A+a
    attn(dPL2M[i], x_pl, SP, mc2, r_pl, EPL, e_pl2m, 64, 1, false);
    attn(dA2M[i], x_a, SA, mc2, r_a, EA, e_a2m, 32, 2, false);
    attn(dM2M[i], mc2, N, mc2, r_a, EA, e_a2m, 32, 2, false);
    perm_kernel<<<ew(N * 128), 256, 0, stream>>>(mc2, mc, S, A);   // -> a*S+s
    attn(dMS2M[i], msv, A * MODES, mc, nullptr, 0, nullptr, MODES, 3, true);
    add_kernel<<<ew(N * 128), 256, 0, stream>>>(msum, mc, msum, N * 128);
  }

  // ---- skip MLP + output projection ----
  gemm(msum, to_wp, to_b, msum, t1, N, 128, 128, ACT_GELU);  // x + gelu(x@w+b)
  ln(t1, to_n1_g, to_n1_b, t2, N);
  add_kernel<<<ew(N * 128), 256, 0, stream>>>(msum, t2, t1, N * 128);
  ln(t1, to_n2_g, to_n2_b, t2, N);
  gemm(t2, proj_out_p, proj_out_b, nullptr, (float*)d_out, N, 32, 128, ACT_NONE);
}
